// SSDKernel_70738111365302
// MI455X (gfx1250) — matrix operand order repaired and checked by reading, not compiled, not run on hardware
//
#include <hip/hip_runtime.h>
#include <hip/hip_bf16.h>

// ---------------------------------------------------------------------------
// Mamba-2 SSD chunked scan for MI455X (gfx1250).
// All GEMMs run on V_WMMA_F32_16X16X32_BF16 (f32 accumulate, bf16 operands).
// Layout convention for every GEMM:  D(M,N) = A(M,K) * Bt(N,K)^T, both
// operands K-contiguous in memory -> all WMMA fragment loads are 16B vectors.
// ---------------------------------------------------------------------------

typedef __attribute__((ext_vector_type(16))) __bf16 bf16x16;
typedef __attribute__((ext_vector_type(8)))  __bf16 bf16x8;
typedef __attribute__((ext_vector_type(8)))  float  f32x8;

#define BATCH   4
#define SEQ     2048
#define DMODEL  192
#define NSTATE  64
#define CHUNK   128
#define NROWS   (BATCH * SEQ)          // 8192
#define NCHUNK  (NROWS / CHUNK)        // 64
#define CPB     (SEQ / CHUNK)          // 16 chunks per batch

// Native bf16 convert: let the backend use the gfx1250 hardware cvt op.
__device__ __forceinline__ __bf16 f2bf(float f) { return static_cast<__bf16>(f); }

__device__ __forceinline__ float fast_rcp(float x) { return __builtin_amdgcn_rcpf(x); }

// A fragment (16x32, row-major, ld elements). ISA layout: lanes 0-15 row=lane,
// elems 0..7 = K0..7, 8..15 = K16..23; lanes 16-31: K8..15 / K24..31.
__device__ __forceinline__ bf16x16 load_a_bf(const __bf16* A, int ld, int m0, int k0) {
  int lane = threadIdx.x & 31;
  int m  = m0 + (lane & 15);
  int kb = k0 + ((lane >> 4) << 3);
  const __bf16* p = A + m * ld + kb;
  bf16x8 lo = *reinterpret_cast<const bf16x8*>(p);
  bf16x8 hi = *reinterpret_cast<const bf16x8*>(p + 16);
  bf16x16 a;
#pragma unroll
  for (int i = 0; i < 8; ++i) { a[i] = lo[i]; a[i + 8] = hi[i]; }
  return a;
}

// Same fragment but from f32 memory (convert on the fly) -- only for input x.
__device__ __forceinline__ bf16x16 load_a_f32(const float* A, int ld, int m0, int k0) {
  int lane = threadIdx.x & 31;
  int m  = m0 + (lane & 15);
  int kb = k0 + ((lane >> 4) << 3);
  const float* p = A + m * ld + kb;
  bf16x16 a;
#pragma unroll
  for (int i = 0; i < 8; ++i) { a[i] = f2bf(p[i]); a[i + 8] = f2bf(p[i + 16]); }
  return a;
}

// B fragment from Bt stored (N,K) row-major: lane col = n0+lane%16,
// elems 0..7 = K k0+(lane/16)*8.., 8..15 = K k0+16+(lane/16)*8..
__device__ __forceinline__ bf16x16 load_bt_bf(const __bf16* Bt, int ld, int n0, int k0) {
  int lane = threadIdx.x & 31;
  int n  = n0 + (lane & 15);
  int kb = k0 + ((lane >> 4) << 3);
  const __bf16* p = Bt + n * ld + kb;
  bf16x8 lo = *reinterpret_cast<const bf16x8*>(p);
  bf16x8 hi = *reinterpret_cast<const bf16x8*>(p + 16);
  bf16x16 b;
#pragma unroll
  for (int i = 0; i < 8; ++i) { b[i] = lo[i]; b[i + 8] = hi[i]; }
  return b;
}

__device__ __forceinline__ f32x8 wmma_bf16(bf16x16 a, bf16x16 b, f32x8 c) {
  return __builtin_amdgcn_wmma_f32_16x16x32_bf16(false, a, false, b,
                                                 (short)0, c, false, false);
}

// ---------------------------------------------------------------------------
__global__ void k_cvt(const float* __restrict__ src, __bf16* __restrict__ dst, int n) {
  int i = blockIdx.x * 256 + threadIdx.x;
  if (i < n) dst[i] = f2bf(src[i]);
}

// proj = x @ W_xp^T + b_xp ; split into x_bar (f32 + bf16) and silu(z) (f32)
// grid(512,3) block 256 (8 waves), tile 16x16 per wave, N = 384
__global__ void k_proj(const float* __restrict__ x, const __bf16* __restrict__ Wxp,
                       const float* __restrict__ bxp,
                       float* __restrict__ xbar_f, __bf16* __restrict__ xbar_b,
                       float* __restrict__ zsilu) {
  int wave = threadIdx.x >> 5, lane = threadIdx.x & 31;
  int m0 = blockIdx.x * 16;
  int n0 = (blockIdx.y * 8 + wave) * 16;
  f32x8 acc = {};
#pragma unroll
  for (int k = 0; k < DMODEL; k += 32)
    acc = wmma_bf16(load_a_f32(x, DMODEL, m0, k), load_bt_bf(Wxp, DMODEL, n0, k), acc);
  int ncol = n0 + (lane & 15);
  int mrow = m0 + ((lane >> 4) << 3);
  float bias = bxp[ncol];
  if (n0 < DMODEL) {           // x_bar half: uniform per wave (tiles 16-aligned)
#pragma unroll
    for (int r = 0; r < 8; ++r) {
      float v = acc[r] + bias;
      int row = mrow + r;
      xbar_f[row * DMODEL + ncol] = v;
      xbar_b[row * DMODEL + ncol] = f2bf(v);
    }
  } else {                     // z half -> silu(z), fast sigmoid via v_rcp_f32
    int zc = ncol - DMODEL;
#pragma unroll
    for (int r = 0; r < 8; ++r) {
      float v = acc[r] + bias;
      zsilu[(mrow + r) * DMODEL + zc] = v * fast_rcp(1.0f + __expf(-v));
    }
  }
}

// BC = x_bar @ [W_B;W_C]^T -> BC (8192,128) bf16; Bm also stored chunk-transposed.
// grid(512) block 256: wave = n-tile (8 tiles of 16 over N=128)
__global__ void k_bc(const __bf16* __restrict__ xbar_b, const __bf16* __restrict__ WB,
                     const __bf16* __restrict__ WC, __bf16* __restrict__ BC,
                     __bf16* __restrict__ BmT) {
  int wave = threadIdx.x >> 5, lane = threadIdx.x & 31;
  int m0 = blockIdx.x * 16;
  int n0 = wave * 16;
  const __bf16* W = (n0 < NSTATE) ? (WB + n0 * DMODEL) : (WC + (n0 - NSTATE) * DMODEL);
  f32x8 acc = {};
#pragma unroll
  for (int k = 0; k < DMODEL; k += 32)
    acc = wmma_bf16(load_a_bf(xbar_b, DMODEL, m0, k), load_bt_bf(W, DMODEL, 0, k), acc);
  int ncol = n0 + (lane & 15);
  int mrow = m0 + ((lane >> 4) << 3);
  bf16x8 vb;
#pragma unroll
  for (int r = 0; r < 8; ++r) {
    vb[r] = f2bf(acc[r]);
    BC[(mrow + r) * 128 + ncol] = vb[r];
  }
  if (n0 < NSTATE) {   // Bm transposed per chunk: (chunk, n=64, p=128)
    int chunk = m0 >> 7;
    int pin = (m0 & 127) + ((lane >> 4) << 3);
    // D-tile is column-resident per lane -> 8 consecutive p's: one 16B store
    *reinterpret_cast<bf16x8*>(BmT + chunk * (NSTATE * CHUNK) + ncol * CHUNK + pin) = vb;
  }
}

// la = -exp(A_log[d]) * softplus(x_bar @ W_dt^T + b_dt)   (f32, 8192x192)
// grid(512,2) block 192 (6 waves)
__global__ void k_dt(const __bf16* __restrict__ xbar_b, const __bf16* __restrict__ Wdt,
                     const float* __restrict__ bdt, const float* __restrict__ A_log,
                     float* __restrict__ la) {
  int wave = threadIdx.x >> 5, lane = threadIdx.x & 31;
  int m0 = blockIdx.x * 16;
  int n0 = (blockIdx.y * 6 + wave) * 16;
  f32x8 acc = {};
#pragma unroll
  for (int k = 0; k < DMODEL; k += 32)
    acc = wmma_bf16(load_a_bf(xbar_b, DMODEL, m0, k), load_bt_bf(Wdt, DMODEL, n0, k), acc);
  int ncol = n0 + (lane & 15);
  int mrow = m0 + ((lane >> 4) << 3);
  float bias = bdt[ncol];
  float Ad = -__expf(A_log[ncol]);
#pragma unroll
  for (int r = 0; r < 8; ++r) {
    float v = acc[r] + bias;
    // softplus on v_exp/v_log transcendentals (fine for epilogue accuracy)
    float sp = (v > 15.f) ? v : __logf(1.0f + __expf(v));
    la[(mrow + r) * DMODEL + ncol] = Ad * sp;
  }
}

// Per (chunk, d): serial cumsum of la over p=128.
// Emits decay=exp(lacum) f32, x_hat^T = (x*exp(-lacum))^T bf16,
// x2^T = (x*exp(la))^T bf16, chunkA = exp(lacum[127]).
// grid(64) block 192
__global__ void k_decay(const float* __restrict__ la, const float* __restrict__ xbar_f,
                        float* __restrict__ decay, __bf16* __restrict__ xhatT,
                        __bf16* __restrict__ x2T, float* __restrict__ chunkA) {
  int chunk = blockIdx.x, d = threadIdx.x;
  const float* lap = la + (size_t)chunk * CHUNK * DMODEL + d;
  const float* xp  = xbar_f + (size_t)chunk * CHUNK * DMODEL + d;
  float* dec = decay + (size_t)chunk * CHUNK * DMODEL + d;
  __bf16* xh = xhatT + (size_t)chunk * DMODEL * CHUNK + d * CHUNK;
  __bf16* x2 = x2T   + (size_t)chunk * DMODEL * CHUNK + d * CHUNK;
  float s = 0.f;
  float es = 1.f;                        // exp(s), recomputed from the running sum s each step
  for (int p = 0; p < CHUNK; ++p) {
    float l = lap[p * DMODEL];
    s += l;
    es = __expf(s);
    float xb = xp[p * DMODEL];
    dec[p * DMODEL] = es;
    xh[p] = f2bf(xb * fast_rcp(es));     // exp(-s) = 1/exp(s)
    x2[p] = f2bf(xb * __expf(l));
  }
  chunkA[chunk * DMODEL + d] = es;
}

// G = Cc @ Bc^T (128x128, K=64), causal-masked, stored bf16.
// grid(8,64) block 256: blockIdx.x = i-tile, wave = j-tile
__global__ void k_g(const __bf16* __restrict__ BC, __bf16* __restrict__ Gm) {
  int wave = threadIdx.x >> 5, lane = threadIdx.x & 31;
  int chunk = blockIdx.y;
  int i0 = blockIdx.x * 16, j0 = wave * 16;
  const __bf16* Cc = BC + (size_t)chunk * CHUNK * 128 + NSTATE;  // (p,64) ld=128
  const __bf16* Bt = BC + (size_t)chunk * CHUNK * 128;           // (p,64) ld=128
  f32x8 acc = {};
#pragma unroll
  for (int k = 0; k < NSTATE; k += 32)
    acc = wmma_bf16(load_a_bf(Cc, 128, i0, k), load_bt_bf(Bt, 128, j0, k), acc);
  int j  = j0 + (lane & 15);
  int ib = i0 + ((lane >> 4) << 3);
  __bf16* dst = Gm + (size_t)chunk * CHUNK * CHUNK;
#pragma unroll
  for (int r = 0; r < 8; ++r) {
    int i = ib + r;
    dst[i * CHUNK + j] = f2bf((j <= i) ? acc[r] : 0.f);
  }
}

// states = Bc^T @ (x*A)  -> (chunk, 64, 192) f32.  A = BmT, Bt = x2T.
// grid(6,64) block 256: wave -> (tm = wave&3, tn = bx*2 + wave>>2)
__global__ void k_states(const __bf16* __restrict__ BmT, const __bf16* __restrict__ x2T,
                         float* __restrict__ states) {
  int wave = threadIdx.x >> 5, lane = threadIdx.x & 31;
  int chunk = blockIdx.y;
  int m0 = (wave & 3) * 16;
  int n0 = (blockIdx.x * 2 + (wave >> 2)) * 16;
  const __bf16* A  = BmT + (size_t)chunk * NSTATE * CHUNK;
  const __bf16* Bt = x2T + (size_t)chunk * DMODEL * CHUNK;
  f32x8 acc = {};
#pragma unroll
  for (int k = 0; k < CHUNK; k += 32)
    acc = wmma_bf16(load_a_bf(A, CHUNK, m0, k), load_bt_bf(Bt, CHUNK, n0, k), acc);
  int ncol = n0 + (lane & 15);
  int mrow = m0 + ((lane >> 4) << 3);
  float* dst = states + (size_t)chunk * NSTATE * DMODEL;
#pragma unroll
  for (int r = 0; r < 8; ++r) dst[(mrow + r) * DMODEL + ncol] = acc[r];
}

// Inter-chunk scan: state(n,d) per batch, 16 sequential chunks, state in regs.
// Emits PRE-update state transposed (chunk, d=192, n=64) bf16.  grid(4) block 256
__global__ void k_scan(const float* __restrict__ states, const float* __restrict__ chunkA,
                       __bf16* __restrict__ prevT) {
  int b = blockIdx.x, t = threadIdx.x;
  float st[48];
#pragma unroll
  for (int k = 0; k < 48; ++k) st[k] = 0.f;
  for (int c = 0; c < CPB; ++c) {
    int chunk = b * CPB + c;
    const float* cs = states + (size_t)chunk * NSTATE * DMODEL;
    const float* ca = chunkA + chunk * DMODEL;
    __bf16* pv = prevT + (size_t)chunk * DMODEL * NSTATE;
#pragma unroll
    for (int k = 0; k < 48; ++k) {
      int idx = t + 256 * k;             // 0..12287
      int nn = idx / DMODEL, dd = idx - nn * DMODEL;
      pv[dd * NSTATE + nn] = f2bf(st[k]);          // emit state BEFORE update
      st[k] = st[k] * ca[dd] + cs[idx];
    }
  }
}

// y = (decay * (G_hat @ x_hat) + Cc @ prev) * silu(z)  -> ybar bf16 (8192,192)
// grid(12,64) block 256: wave = i-tile (8), blockIdx.x = d-tile (12)
__global__ void k_y(const __bf16* __restrict__ Gm, const __bf16* __restrict__ xhatT,
                    const __bf16* __restrict__ BC, const __bf16* __restrict__ prevT,
                    const float* __restrict__ decay, const float* __restrict__ zsilu,
                    __bf16* __restrict__ ybar) {
  int wave = threadIdx.x >> 5, lane = threadIdx.x & 31;
  int chunk = blockIdx.y;
  int i0 = wave * 16, n0 = blockIdx.x * 16;
  const __bf16* G  = Gm + (size_t)chunk * CHUNK * CHUNK;
  const __bf16* Xh = xhatT + (size_t)chunk * DMODEL * CHUNK;
  const __bf16* Cc = BC + (size_t)chunk * CHUNK * 128 + NSTATE;
  const __bf16* Pv = prevT + (size_t)chunk * DMODEL * NSTATE;
  f32x8 acc = {};
#pragma unroll
  for (int k = 0; k < CHUNK; k += 32)      // intra: (128x128)@(128x192)
    acc = wmma_bf16(load_a_bf(G, CHUNK, i0, k), load_bt_bf(Xh, CHUNK, n0, k), acc);
  f32x8 acc2 = {};
#pragma unroll
  for (int k = 0; k < NSTATE; k += 32)     // inter: (128x64)@(64x192)
    acc2 = wmma_bf16(load_a_bf(Cc, 128, i0, k), load_bt_bf(Pv, NSTATE, n0, k), acc2);
  int ncol = n0 + (lane & 15);
  int ib = i0 + ((lane >> 4) << 3);
#pragma unroll
  for (int r = 0; r < 8; ++r) {
    int row = chunk * CHUNK + ib + r;
    float y = acc[r] * decay[row * DMODEL + ncol] + acc2[r];
    ybar[row * DMODEL + ncol] = f2bf(y * zsilu[row * DMODEL + ncol]);
  }
}

// out = ybar @ W_out^T + b_out  (f32).  grid(512,2) block 192
__global__ void k_out(const __bf16* __restrict__ ybar, const __bf16* __restrict__ Wout,
                      const float* __restrict__ bout, float* __restrict__ out) {
  int wave = threadIdx.x >> 5, lane = threadIdx.x & 31;
  int m0 = blockIdx.x * 16;
  int n0 = (blockIdx.y * 6 + wave) * 16;
  f32x8 acc = {};
#pragma unroll
  for (int k = 0; k < DMODEL; k += 32)
    acc = wmma_bf16(load_a_bf(ybar, DMODEL, m0, k), load_bt_bf(Wout, DMODEL, n0, k), acc);
  int ncol = n0 + (lane & 15);
  int mrow = m0 + ((lane >> 4) << 3);
  float bias = bout[ncol];
#pragma unroll
  for (int r = 0; r < 8; ++r) out[(mrow + r) * DMODEL + ncol] = acc[r] + bias;
}

// ---------------------------------------------------------------------------
extern "C" void kernel_launch(void* const* d_in, const int* in_sizes, int n_in,
                              void* d_out, int out_size, void* d_ws, size_t ws_size,
                              hipStream_t stream) {
  const float* x     = (const float*)d_in[0];
  const float* A_log = (const float*)d_in[1];
  const float* W_B   = (const float*)d_in[2];
  const float* W_C   = (const float*)d_in[3];
  const float* W_dt  = (const float*)d_in[4];
  const float* b_dt  = (const float*)d_in[5];
  const float* W_xp  = (const float*)d_in[6];
  const float* b_xp  = (const float*)d_in[7];
  const float* W_out = (const float*)d_in[8];
  const float* b_out = (const float*)d_in[9];
  float* out = (float*)d_out;

  char* p = (char*)d_ws;
  auto alloc = [&](size_t bytes) -> char* {
    char* r = p; p += (bytes + 255) & ~size_t(255); return r;
  };
  __bf16* wbB   = (__bf16*)alloc((size_t)NSTATE * DMODEL * 2);
  __bf16* wbC   = (__bf16*)alloc((size_t)NSTATE * DMODEL * 2);
  __bf16* wbdt  = (__bf16*)alloc((size_t)DMODEL * DMODEL * 2);
  __bf16* wbxp  = (__bf16*)alloc((size_t)2 * DMODEL * DMODEL * 2);
  __bf16* wbout = (__bf16*)alloc((size_t)DMODEL * DMODEL * 2);
  float*  xbar_f = (float*) alloc((size_t)NROWS * DMODEL * 4);
  __bf16* xbar_b = (__bf16*)alloc((size_t)NROWS * DMODEL * 2);
  float*  zsilu  = (float*) alloc((size_t)NROWS * DMODEL * 4);
  __bf16* BC     = (__bf16*)alloc((size_t)NROWS * 128 * 2);
  __bf16* BmT    = (__bf16*)alloc((size_t)NCHUNK * NSTATE * CHUNK * 2);
  float*  la     = (float*) alloc((size_t)NROWS * DMODEL * 4);
  float*  decay  = (float*) alloc((size_t)NROWS * DMODEL * 4);
  __bf16* xhatT  = (__bf16*)alloc((size_t)NCHUNK * DMODEL * CHUNK * 2);
  __bf16* x2T    = (__bf16*)alloc((size_t)NCHUNK * DMODEL * CHUNK * 2);
  float*  chunkA = (float*) alloc((size_t)NCHUNK * DMODEL * 4);
  __bf16* Gm     = (__bf16*)alloc((size_t)NCHUNK * CHUNK * CHUNK * 2);
  float*  states = (float*) alloc((size_t)NCHUNK * NSTATE * DMODEL * 4);
  __bf16* prevT  = (__bf16*)alloc((size_t)NCHUNK * DMODEL * NSTATE * 2);
  __bf16* ybar   = (__bf16*)alloc((size_t)NROWS * DMODEL * 2);
  (void)ws_size; (void)in_sizes; (void)n_in; (void)out_size;

  auto cvt = [&](const float* s, __bf16* d, int n) {
    k_cvt<<<dim3((n + 255) / 256), dim3(256), 0, stream>>>(s, d, n);
  };
  cvt(W_B,   wbB,   NSTATE * DMODEL);
  cvt(W_C,   wbC,   NSTATE * DMODEL);
  cvt(W_dt,  wbdt,  DMODEL * DMODEL);
  cvt(W_xp,  wbxp,  2 * DMODEL * DMODEL);
  cvt(W_out, wbout, DMODEL * DMODEL);

  k_proj  <<<dim3(NROWS / 16, 3), dim3(256), 0, stream>>>(x, wbxp, b_xp, xbar_f, xbar_b, zsilu);
  k_bc    <<<dim3(NROWS / 16),    dim3(256), 0, stream>>>(xbar_b, wbB, wbC, BC, BmT);
  k_dt    <<<dim3(NROWS / 16, 2), dim3(192), 0, stream>>>(xbar_b, wbdt, b_dt, A_log, la);
  k_decay <<<dim3(NCHUNK),        dim3(192), 0, stream>>>(la, xbar_f, decay, xhatT, x2T, chunkA);
  k_g     <<<dim3(8, NCHUNK),     dim3(256), 0, stream>>>(BC, Gm);
  k_states<<<dim3(6, NCHUNK),     dim3(256), 0, stream>>>(BmT, x2T, states);
  k_scan  <<<dim3(BATCH),         dim3(256), 0, stream>>>(states, chunkA, prevT);
  k_y     <<<dim3(12, NCHUNK),    dim3(256), 0, stream>>>(Gm, xhatT, BC, prevT, decay, zsilu, ybar);
  k_out   <<<dim3(NROWS / 16, 2), dim3(192), 0, stream>>>(ybar, wbout, b_out, out);
}
